// GEncOpt_56401510531402
// MI455X (gfx1250) — matrix operand order repaired and checked by reading, not compiled, not run on hardware
//
#include <hip/hip_runtime.h>

// ---------------------------------------------------------------------------
// Graph VAE encoder (GCN x3 + reparam) for MI455X / gfx1250 (wave32, WMMA).
// GEMMs: v_wmma_f32_16x16x32_bf16, operands pre-packed into the ISA's wave32
// fragment layout (32B contiguous per lane per fragment -> two
// global_load_b128, fully coalesced). Wave index forced into SGPRs via
// readfirstlane so all control flow is scalar and EXEC stays all-ones for
// WMMA. Graph gather/scatter stays f32; atomics resolve in the 192MB L2.
// ---------------------------------------------------------------------------

typedef __attribute__((ext_vector_type(16))) __bf16          v16bf;
typedef __attribute__((ext_vector_type(8)))  float           v8f;
typedef __attribute__((ext_vector_type(8)))  unsigned short  v8us;

union frag_u { v8us h[2]; v16bf v; };

__device__ __forceinline__ unsigned short f32_to_bf16_rne(float f) {
    unsigned int u = __float_as_uint(f);
    u += 0x7FFFu + ((u >> 16) & 1u);          // round-to-nearest-even
    return (unsigned short)(u >> 16);
}

// ---------------- graph / elementwise kernels ----------------

__global__ void k_set1(float* __restrict__ p, int n) {
    int i = blockIdx.x * blockDim.x + threadIdx.x;
    if (i < n) p[i] = 1.0f;                    // deg starts at 1 (self-loop)
}

__global__ void k_deg(const int* __restrict__ dst, float* __restrict__ deg, int e) {
    int i = blockIdx.x * blockDim.x + threadIdx.x;
    if (i < e) atomicAdd(&deg[dst[i]], 1.0f);
}

__global__ void k_rsqrt_inplace(float* __restrict__ p, int n) {
    int i = blockIdx.x * blockDim.x + threadIdx.x;
    if (i < n) p[i] = rsqrtf(p[i]);            // deg >= 1 always
}

// out[node][f] = dinv[node]^2 * t[node][f] + bias[f]   (self-loop + bias init)
template<int F>
__global__ void k_selfinit(const float* __restrict__ dinv,
                           const float* __restrict__ t,
                           const float* __restrict__ bias,
                           float* __restrict__ out, int n) {
    unsigned i = blockIdx.x * blockDim.x + threadIdx.x;   // < n*F <= 12.8M
    if (i >= (unsigned)n * F) return;
    int node = i / F;                                     // F is constexpr pow2
    int f    = i & (F - 1);
    float dv = dinv[node];
    out[i] = dv * dv * t[i] + bias[f];
}

// per-edge normalized scatter-add: out[dst] += dinv[src]*dinv[dst] * t[src]
// one thread per (edge, 4-float chunk); f32 atomics resolve in L2.
template<int F>
__global__ void k_edgeagg(const int* __restrict__ src,
                          const int* __restrict__ dst,
                          const float* __restrict__ dinv,
                          const float* __restrict__ t,
                          float* __restrict__ out, int e) {
    constexpr int F4 = F >> 2;
    long long i = (long long)blockIdx.x * blockDim.x + threadIdx.x;
    if (i >= (long long)e * F4) return;
    int edge = (int)(i / F4);
    int c    = ((int)i & (F4 - 1)) << 2;
    int s = src[edge];
    int d = dst[edge];
    float nrm = dinv[s] * dinv[d];
    const float4 v = *(const float4*)(t + (unsigned)s * F + c);
    float* o = out + (unsigned)d * F + c;
    atomicAdd(o + 0, nrm * v.x);
    atomicAdd(o + 1, nrm * v.y);
    atomicAdd(o + 2, nrm * v.z);
    atomicAdd(o + 3, nrm * v.w);
}

__global__ void k_reparam(const float* __restrict__ init_dist,
                          const float* __restrict__ ls,
                          float* __restrict__ out, unsigned n) {
    unsigned i = blockIdx.x * blockDim.x + threadIdx.x;
    if (i < n) out[i] += init_dist[i] * __expf(ls[i]);
}

// ---------------- fragment packing (f32 -> bf16, fragment-native) ----------
// Packed layout: fragment index fr owns 32 lanes x 16 bf16 (1KB):
//   element address = (fr*32 + lane)*16 + j
//
// A (16x32 per fragment, ISA 7.12.2): fr = mtile*(K/32) + ks
//   lane: m = mtile*16 + (lane&15), kb = 8*(lane>>4)
//   j:    k = ks*32 + kb + (j&7) + ((j>>3)<<4)
__global__ void k_pack_a(const float* __restrict__ X, unsigned short* __restrict__ out,
                         int M, int K) {
    unsigned tid = blockIdx.x * blockDim.x + threadIdx.x;
    if (tid >= (unsigned)M * K) return;
    const int KS   = K >> 5;
    int j     = tid & 15;
    int lane  = (tid >> 4) & 31;
    int fr    = tid >> 9;
    int ks    = fr % KS;
    int mtile = fr / KS;
    int m  = (mtile << 4) + (lane & 15);
    int kb = (lane >> 4) << 3;
    int k  = (ks << 5) + kb + (j & 7) + ((j >> 3) << 4);
    out[tid] = f32_to_bf16_rne(X[(unsigned)m * K + k]);
}

// B (32x16 per fragment): fr = ntile*(K/32) + ks
//   lane: n = ntile*16 + (lane&15)
//   j:    k = ks*32 + 8*(lane>>4) + (j&7) + ((j>>3)<<4)
__global__ void k_pack_w(const float* __restrict__ W, unsigned short* __restrict__ out,
                         int K, int F) {
    unsigned tid = blockIdx.x * blockDim.x + threadIdx.x;
    if (tid >= (unsigned)K * F) return;
    const int KS = K >> 5;
    int j     = tid & 15;
    int lane  = (tid >> 4) & 31;
    int fr    = tid >> 9;
    int ks    = fr % KS;
    int ntile = fr / KS;
    int k = (ks << 5) + ((lane >> 4) << 3) + (j & 7) + ((j >> 3) << 4);
    int n = (ntile << 4) + (lane & 15);
    out[tid] = f32_to_bf16_rne(W[(unsigned)k * F + n]);
}

// ---------------- WMMA GEMM (packed operands, no divergent guards) ---------
// D[f32 MxF] = A @ B.  One wave handles MT consecutive 16-row tiles for one
// 16-col tile; B fragment loaded once per k-step, reused MT times.
// Host guarantees: every launched wave with mgrp < mgrps is fully in-bounds.
// C/D layout: lane n=l%16, vgpr r -> row = 16*mtile + r + 8*(l/16).
template<int K, int MT>
__global__ void k_wmma_gemm(const unsigned short* __restrict__ Ap,
                            const unsigned short* __restrict__ Bp,
                            float* __restrict__ D,
                            int mtile0, int mgrps, int F) {
    constexpr int KS = K >> 5;
    // wave-uniform scalars (readfirstlane -> SGPR, scalar branches, saddr loads)
    const int wave = __builtin_amdgcn_readfirstlane(
        blockIdx.x * (blockDim.x >> 5) + (threadIdx.x >> 5));
    const int lane = threadIdx.x & 31;
    const int ntiles = F >> 4;
    const int mgrp  = wave / ntiles;
    const int ntile = wave - mgrp * ntiles;
    if (mgrp >= mgrps) return;                 // scalar branch, EXEC untouched
    const int mtileB = mtile0 + mgrp * MT;

    const unsigned short* bbase = Bp + ((unsigned)(ntile * KS) * 32 + lane) * 16;
    const unsigned short* abase = Ap + ((unsigned)(mtileB * KS) * 32 + lane) * 16;

    v8f acc[MT] = {};
#pragma unroll
    for (int ks = 0; ks < KS; ++ks) {
        frag_u b;
        b.h[0] = *(const v8us*)(bbase + ks * 512);
        b.h[1] = *(const v8us*)(bbase + ks * 512 + 8);
#pragma unroll
        for (int t = 0; t < MT; ++t) {
            frag_u a;
            const unsigned short* p = abase + (unsigned)(t * KS + ks) * 512;
            a.h[0] = *(const v8us*)p;
            a.h[1] = *(const v8us*)(p + 8);
            acc[t] = __builtin_amdgcn_wmma_f32_16x16x32_bf16(
                         false, a.v, false, b.v, (short)0, acc[t], false, false);
        }
    }
    const int nc = (ntile << 4) + (lane & 15);
    const int r0 = (lane >> 4) << 3;
#pragma unroll
    for (int t = 0; t < MT; ++t)
#pragma unroll
        for (int r = 0; r < 8; ++r)
            D[(unsigned)((mtileB + t) * 16 + r0 + r) * F + nc] = acc[t][r];
}

// Dual-B variant: one A stream feeds two weight matrices (mu / logstd).
template<int K, int MT>
__global__ void k_wmma_gemm2(const unsigned short* __restrict__ Ap,
                             const unsigned short* __restrict__ Bp0,
                             const unsigned short* __restrict__ Bp1,
                             float* __restrict__ D0,
                             float* __restrict__ D1,
                             int mtile0, int mgrps, int F) {
    constexpr int KS = K >> 5;
    const int wave = __builtin_amdgcn_readfirstlane(
        blockIdx.x * (blockDim.x >> 5) + (threadIdx.x >> 5));
    const int lane = threadIdx.x & 31;
    const int ntiles = F >> 4;
    const int mgrp  = wave / ntiles;
    const int ntile = wave - mgrp * ntiles;
    if (mgrp >= mgrps) return;                 // scalar branch
    const int mtileB = mtile0 + mgrp * MT;

    const unsigned bo = ((unsigned)(ntile * KS) * 32 + lane) * 16;
    const unsigned short* abase = Ap + ((unsigned)(mtileB * KS) * 32 + lane) * 16;

    v8f acc0[MT] = {};
    v8f acc1[MT] = {};
#pragma unroll
    for (int ks = 0; ks < KS; ++ks) {
        frag_u b0, b1;
        b0.h[0] = *(const v8us*)(Bp0 + bo + ks * 512);
        b0.h[1] = *(const v8us*)(Bp0 + bo + ks * 512 + 8);
        b1.h[0] = *(const v8us*)(Bp1 + bo + ks * 512);
        b1.h[1] = *(const v8us*)(Bp1 + bo + ks * 512 + 8);
#pragma unroll
        for (int t = 0; t < MT; ++t) {
            frag_u a;
            const unsigned short* p = abase + (unsigned)(t * KS + ks) * 512;
            a.h[0] = *(const v8us*)p;
            a.h[1] = *(const v8us*)(p + 8);
            acc0[t] = __builtin_amdgcn_wmma_f32_16x16x32_bf16(
                          false, a.v, false, b0.v, (short)0, acc0[t], false, false);
            acc1[t] = __builtin_amdgcn_wmma_f32_16x16x32_bf16(
                          false, a.v, false, b1.v, (short)0, acc1[t], false, false);
        }
    }
    const int nc = (ntile << 4) + (lane & 15);
    const int r0 = (lane >> 4) << 3;
#pragma unroll
    for (int t = 0; t < MT; ++t)
#pragma unroll
        for (int r = 0; r < 8; ++r) {
            unsigned o = (unsigned)((mtileB + t) * 16 + r0 + r) * F + nc;
            D0[o] = acc0[t][r];
            D1[o] = acc1[t][r];
        }
}

// ---------------------------------------------------------------------------

extern "C" void kernel_launch(void* const* d_in, const int* in_sizes, int n_in,
                              void* d_out, int out_size, void* d_ws, size_t ws_size,
                              hipStream_t stream) {
    constexpr int IN  = 128;
    constexpr int HID = 128;
    constexpr int OUT = 64;
    constexpr int MT  = 5;                     // 6250 m-tiles = 1250 groups, exact

    const float* x    = (const float*)d_in[0];
    const int*   ei   = (const int*)  d_in[1];
    const float* init = (const float*)d_in[2];
    const float* W1   = (const float*)d_in[3];
    const float* b1   = (const float*)d_in[4];
    const float* Wmu  = (const float*)d_in[5];
    const float* bmu  = (const float*)d_in[6];
    const float* Wls  = (const float*)d_in[7];
    const float* bls  = (const float*)d_in[8];
    float* out = (float*)d_out;

    const int N = in_sizes[0] / IN;            // 100000 (divisible by 16)
    const int E = in_sizes[1] / 2;             // 1600000
    const int* src = ei;
    const int* dst = ei + E;
    const int Mtiles = N / 16;                 // 6250
    const int mgrps  = Mtiles / MT;            // full groups
    const int tail   = Mtiles - mgrps * MT;    // leftover tiles (0 here)

    // ---- workspace carve-out (256B aligned) ----
    char* wsp = (char*)d_ws;
    size_t off = 0;
    auto alloc = [&](size_t bytes) -> void* {
        void* p = wsp + off;
        off = (off + bytes + 255) & ~(size_t)255;
        return p;
    };
    float*          dinv = (float*)         alloc((size_t)N * 4);          // deg -> dinv
    unsigned short* ap   = (unsigned short*)alloc((size_t)N * HID * 2);    // packed A (x, then h1)
    unsigned short* wb1  = (unsigned short*)alloc((size_t)IN * HID * 2);   // packed weights
    unsigned short* wbmu = (unsigned short*)alloc((size_t)HID * OUT * 2);
    unsigned short* wbls = (unsigned short*)alloc((size_t)HID * OUT * 2);
    float*          t1   = (float*)         alloc((size_t)N * HID * 4);    // reused: tmu | tls
    float*          h1   = (float*)         alloc((size_t)N * HID * 4);    // reused: ls
    float* tmu = t1;
    float* tls = t1 + (size_t)N * OUT;
    float* ls  = h1;
    (void)ws_size; (void)n_in; (void)out_size;

    const int B256 = 256;
    auto blocks = [](long long tot, int b) { return (int)((tot + b - 1) / b); };

    // 1) degree (A + I) and dinv = rsqrt(deg)
    k_set1<<<blocks(N, B256), B256, 0, stream>>>(dinv, N);
    k_deg<<<blocks(E, B256), B256, 0, stream>>>(dst, dinv, E);
    k_rsqrt_inplace<<<blocks(N, B256), B256, 0, stream>>>(dinv, N);

    // 2) pack operands to bf16 fragment layout
    k_pack_a<<<blocks((long long)N * IN, B256), B256, 0, stream>>>(x, ap, N, IN);
    k_pack_w<<<blocks(IN * HID, B256), B256, 0, stream>>>(W1, wb1, IN, HID);
    k_pack_w<<<blocks(HID * OUT, B256), B256, 0, stream>>>(Wmu, wbmu, HID, OUT);
    k_pack_w<<<blocks(HID * OUT, B256), B256, 0, stream>>>(Wls, wbls, HID, OUT);

    // 3) layer 1: t1 = X @ W1 ; h1 = selfloop + scatter-add + b1
    {
        long long waves = (long long)mgrps * (HID / 16);
        k_wmma_gemm<IN, MT><<<blocks(waves, 8), B256, 0, stream>>>(ap, wb1, t1, 0, mgrps, HID);
        if (tail) {
            long long twaves = (long long)tail * (HID / 16);
            k_wmma_gemm<IN, 1><<<blocks(twaves, 8), B256, 0, stream>>>(
                ap, wb1, t1, mgrps * MT, tail, HID);
        }
    }
    k_selfinit<HID><<<blocks((long long)N * HID, B256), B256, 0, stream>>>(dinv, t1, b1, h1, N);
    k_edgeagg<HID><<<blocks((long long)E * (HID / 4), B256), B256, 0, stream>>>(src, dst, dinv, t1, h1, E);

    // 4) h1 -> packed bf16 fragments (reuse ap)
    k_pack_a<<<blocks((long long)N * HID, B256), B256, 0, stream>>>(h1, ap, N, HID);

    // 5) mu / logstd transforms fused: one A stream, two weight matrices
    {
        long long waves = (long long)mgrps * (OUT / 16);
        k_wmma_gemm2<HID, MT><<<blocks(waves, 8), B256, 0, stream>>>(
            ap, wbmu, wbls, tmu, tls, 0, mgrps, OUT);
        if (tail) {
            long long twaves = (long long)tail * (OUT / 16);
            k_wmma_gemm2<HID, 1><<<blocks(twaves, 8), B256, 0, stream>>>(
                ap, wbmu, wbls, tmu, tls, mgrps * MT, tail, OUT);
        }
    }

    // 6) aggregate: mu directly into d_out, logstd into ls
    k_selfinit<OUT><<<blocks((long long)N * OUT, B256), B256, 0, stream>>>(dinv, tmu, bmu, out, N);
    k_selfinit<OUT><<<blocks((long long)N * OUT, B256), B256, 0, stream>>>(dinv, tls, bls, ls, N);
    k_edgeagg<OUT><<<blocks((long long)E * (OUT / 4), B256), B256, 0, stream>>>(src, dst, dinv, tmu, out, E);
    k_edgeagg<OUT><<<blocks((long long)E * (OUT / 4), B256), B256, 0, stream>>>(src, dst, dinv, tls, ls, E);

    // 7) reparametrize: out = mu + init * exp(logstd)
    k_reparam<<<blocks((long long)N * OUT, B256), B256, 0, stream>>>(init, ls, out, (unsigned)(N * OUT));
}
